// MultiheadSelfAttentionVarSizedKeysElementReduce_73177652789999
// MI455X (gfx1250) — matrix operand order repaired and checked by reading, not compiled, not run on hardware
//
#include <hip/hip_runtime.h>
#include <hip/hip_bf16.h>

typedef __attribute__((ext_vector_type(16))) __bf16 v16bf;
typedef __attribute__((ext_vector_type(8)))  __bf16 v8bf;
typedef __attribute__((ext_vector_type(8)))  float  v8f;

#define DDIM 256      // input feature dim D
#define HTOT 256      // total hidden (q / pooled) dim
#define NH   8        // heads
#define HD   32       // head dim
#define KVW  512      // kv projection width (2*H)
#define LDA  264      // padded LDS row stride for bf16 A tiles (breaks bank conflicts)
#define KVWP 516      // padded LDS row stride for f32 kv tile

union ABf { v16bf v; v8bf h[2]; };

// A fragment (16x32 bf16) from LDS row-major [16][LDA].
// Per ISA 7.12.2: lane M = lane&15; lanes 0-15: half0 -> K=kb+0..7, half1 -> K=kb+16..23;
// lanes 16-31: half0 -> K=kb+8..15, half1 -> K=kb+24..31.
__device__ __forceinline__ v16bf load_a_frag(const __bf16* lds, int kb, int lane) {
    int m = lane & 15, g = lane >> 4;
    ABf a;
    a.h[0] = *(const v8bf*)(lds + m * LDA + kb + 8 * g);
    a.h[1] = *(const v8bf*)(lds + m * LDA + kb + 16 + 8 * g);
    return a.v;
}

// B fragment (32x16 bf16) from global Bt stored [Ncols][K] (transposed weights).
// Lane holds column n; lanes 0-15: half0 -> K=kb+0..7, half1 -> K=kb+16..23;
// lanes 16-31: half0 -> K=kb+8..15, half1 -> K=kb+24..31.
__device__ __forceinline__ v16bf load_b_frag(const __bf16* bt, int n, int kb, int lane) {
    int g = lane >> 4;
    ABf b;
    b.h[0] = *(const v8bf*)(bt + (size_t)n * DDIM + kb + 8 * g);
    b.h[1] = *(const v8bf*)(bt + (size_t)n * DDIM + kb + 16 + 8 * g);
    return b.v;
}

// ---------------- prep: convert + transpose weights to bf16 [N][K] ----------------
__global__ void prep_weights(const float* __restrict__ Wq, const float* __restrict__ Wkv,
                             const float* __restrict__ Wo,
                             __bf16* __restrict__ wq_t, __bf16* __restrict__ wkv_t,
                             __bf16* __restrict__ wo_t) {
    int t = blockIdx.x * blockDim.x + threadIdx.x;
    if (t < DDIM * HTOT) {
        int k = t / HTOT, n = t % HTOT;
        wq_t[(size_t)n * DDIM + k] = (__bf16)Wq[t];
        wo_t[(size_t)n * DDIM + k] = (__bf16)Wo[t];
    }
    if (t < DDIM * KVW) {
        int k = t / KVW, n = t % KVW;
        wkv_t[(size_t)n * DDIM + k] = (__bf16)Wkv[t];
    }
}

// ---------------- generic 256-wide GEMM: C[M][256] = A[M][256] @ Bt^T + bias ----------------
// One block = 16-row strip (M is a multiple of 16), 8 waves x 2 N-tiles, K-loop 8 x WMMA bf16.
__global__ void gemm_bf16(const float* __restrict__ A, const __bf16* __restrict__ Bt,
                          const float* __restrict__ bias, float* __restrict__ C) {
    __shared__ __bf16 a_lds[16 * LDA];
    int t = threadIdx.x;
    int strip = blockIdx.x;
    for (int i = 0; i < 16; ++i) {
        float v = A[(size_t)(strip * 16 + i) * DDIM + t];
        a_lds[i * LDA + t] = (__bf16)v;
    }
    __syncthreads();
    int wave = t >> 5, lane = t & 31, g = lane >> 4;
    for (int i = 0; i < 2; ++i) {
        int tile = wave * 2 + i;
        int n = tile * 16 + (lane & 15);
        v8f acc = {};
        for (int kk = 0; kk < DDIM; kk += 32) {
            v16bf a = load_a_frag(a_lds, kk, lane);
            v16bf b = load_b_frag(Bt, n, kk, lane);
            acc = __builtin_amdgcn_wmma_f32_16x16x32_bf16(false, a, false, b,
                                                          (short)0, acc, false, false);
        }
        float bv = bias[n];
        int rowbase = strip * 16 + 8 * g;
        for (int r = 0; r < 8; ++r)
            C[(size_t)(rowbase + r) * HTOT + n] = acc[r] + bv;
    }
}

// ---------------- fused KV-projection + segment softmax + V pooling ----------------
// One block (256 thr / 8 waves) per sample; elements are contiguous (idx sorted).
__global__ void attn_pool(const float* __restrict__ emb, const int* __restrict__ idx,
                          const float* __restrict__ qproj, const __bf16* __restrict__ wkv_t,
                          const float* __restrict__ bkv, float* __restrict__ pooled, int N) {
    __shared__ __bf16 a_lds[16 * LDA];    // emb chunk (bf16), padded stride
    __shared__ float  kv_lds[16 * KVWP];  // k|v chunk (f32), padded stride  ~33 KB
    __shared__ float  q_lds[HTOT];        // this sample's q
    __shared__ float  sc_lds[16 * NH];    // chunk scores
    __shared__ int    seg[2];

    int s = blockIdx.x;
    int t = threadIdx.x;
    if (t == 0) {
        int lo = 0, hi = N;
        while (lo < hi) { int mid = (lo + hi) >> 1; if (idx[mid] < s) lo = mid + 1; else hi = mid; }
        seg[0] = lo;
        hi = N;
        while (lo < hi) { int mid = (lo + hi) >> 1; if (idx[mid] <= s) lo = mid + 1; else hi = mid; }
        seg[1] = lo;
    }
    q_lds[t] = qproj[(size_t)s * HTOT + t];
    __syncthreads();
    int start = seg[0], end = seg[1];

    int wave = t >> 5, lane = t & 31, g = lane >> 4;
    int h = t >> 5, d = t & 31;                   // thread owns (head h, dim d) accumulator
    const float scale = 0.17677669529663688f;     // 1/sqrt(32)
    float m = -__builtin_inff(), l = 0.f, acc = 0.f;

    for (int base = start; base < end; base += 16) {
        int cnt = end - base; if (cnt > 16) cnt = 16;
        // stage emb chunk -> LDS bf16 (zero-pad short chunks)
        for (int i = 0; i < 16; ++i) {
            float v = (i < cnt) ? emb[(size_t)(base + i) * DDIM + t] : 0.f;
            a_lds[i * LDA + t] = (__bf16)v;
        }
        // prefetch next chunk: 256 threads cover the 16KB tile at 64B granularity
        int nb = base + 16;
        if (nb < end) {
            int prow = nb + (t >> 4); if (prow >= end) prow = end - 1;
            __builtin_prefetch(emb + (size_t)prow * DDIM + ((t & 15) << 4), 0, 1);
        }
        __syncthreads();
        // KV projection: 16 x 256 @ 256 x 512, 8 waves x 4 N-tiles x 8 WMMAs
        for (int i = 0; i < 4; ++i) {
            int tile = wave + 8 * i;
            int n = tile * 16 + (lane & 15);
            v8f c = {};
            for (int kk = 0; kk < DDIM; kk += 32) {
                v16bf a = load_a_frag(a_lds, kk, lane);
                v16bf b = load_b_frag(wkv_t, n, kk, lane);
                c = __builtin_amdgcn_wmma_f32_16x16x32_bf16(false, a, false, b,
                                                            (short)0, c, false, false);
            }
            float bv = bkv[n];
            for (int r = 0; r < 8; ++r)
                kv_lds[(r + 8 * g) * KVWP + n] = c[r] + bv;
        }
        __syncthreads();
        // scores: 16 elements x 8 heads, dot of length 32
        if (t < 128) {
            int e = t >> 3, hh = t & 7;
            float sum = 0.f;
            for (int dd = 0; dd < HD; ++dd)
                sum += q_lds[hh * HD + dd] * kv_lds[e * KVWP + hh * HD + dd];
            sc_lds[e * NH + hh] = (e < cnt) ? sum * scale : -__builtin_inff();
        }
        __syncthreads();
        // online softmax + weighted-V accumulate (thread owns one (h,d))
        float cmax = -__builtin_inff();
        for (int e = 0; e < 16; ++e) cmax = fmaxf(cmax, sc_lds[e * NH + h]);
        float newm = fmaxf(m, cmax);
        float corr = __expf(m - newm);
        float lsum = 0.f, asum = 0.f;
        for (int e = 0; e < 16; ++e) {
            float p = __expf(sc_lds[e * NH + h] - newm);
            lsum += p;
            asum += p * kv_lds[e * KVWP + HTOT + h * HD + d];
        }
        m = newm;
        l = l * corr + lsum;
        acc = acc * corr + asum;
        __syncthreads();
    }
    pooled[(size_t)s * HTOT + t] = (l > 0.f) ? (acc / l) : 0.f;
}

extern "C" void kernel_launch(void* const* d_in, const int* in_sizes, int n_in,
                              void* d_out, int out_size, void* d_ws, size_t ws_size,
                              hipStream_t stream) {
    (void)n_in; (void)out_size; (void)ws_size;
    const float* queries = (const float*)d_in[0];
    const float* emb     = (const float*)d_in[1];
    const int*   idx     = (const int*)d_in[2];
    // d_in[3] = num_samples scalar (derived from sizes instead)
    const float* Wq  = (const float*)d_in[4];
    const float* bq  = (const float*)d_in[5];
    const float* Wkv = (const float*)d_in[6];
    const float* bkv = (const float*)d_in[7];
    const float* Wo  = (const float*)d_in[8];
    const float* bo  = (const float*)d_in[9];
    float* out = (float*)d_out;

    int S = in_sizes[0] / DDIM;   // 4096
    int N = in_sizes[2];          // 262144

    char* ws = (char*)d_ws;
    __bf16* wq_t  = (__bf16*)ws;  ws += (size_t)HTOT * DDIM * sizeof(__bf16);
    __bf16* wo_t  = (__bf16*)ws;  ws += (size_t)HTOT * DDIM * sizeof(__bf16);
    __bf16* wkv_t = (__bf16*)ws;  ws += (size_t)KVW  * DDIM * sizeof(__bf16);
    float*  qproj = (float*)ws;   ws += (size_t)S * HTOT * sizeof(float);
    float*  pooled = (float*)ws;  ws += (size_t)S * HTOT * sizeof(float);

    prep_weights<<<(DDIM * KVW + 255) / 256, 256, 0, stream>>>(Wq, Wkv, Wo, wq_t, wkv_t, wo_t);
    gemm_bf16<<<S / 16, 256, 0, stream>>>(queries, wq_t, bq, qproj);
    attn_pool<<<S, 256, 0, stream>>>(emb, idx, qproj, wkv_t, bkv, pooled, N);
    gemm_bf16<<<S / 16, 256, 0, stream>>>(pooled, wo_t, bo, out);
}
